// AutoInt_43937515438532
// MI455X (gfx1250) — matrix operand order repaired and checked by reading, not compiled, not run on hardware
//
#include <hip/hip_runtime.h>
#include <hip/hip_bf16.h>
#include <math.h>
#include <stdint.h>

typedef __attribute__((ext_vector_type(16))) _Float16 v16h;
typedef __attribute__((ext_vector_type(8)))  _Float16 v8h;
typedef __attribute__((ext_vector_type(8)))  float    v8f;
typedef __attribute__((ext_vector_type(4)))  unsigned int u32x4;
typedef __attribute__((ext_vector_type(4)))  int      i32x4;
typedef __attribute__((ext_vector_type(8)))  int      i32x8;

#define LDW 136   // row stride (halves) for 128-col f16 LDS tiles (16B-aligned rows)
#define VLD 40    // row stride (halves) for 32-col f16 LDS tiles

// ---------------------------------------------------------------------------
// WMMA helpers: CDNA5 v_wmma_f32_16x16x32_f16 fragment layouts (ISA 7.12.2)
// A (16x32 f16): lane l -> row m=l&15; halves [k0+sel*8, +8) and [k0+16+sel*8, +8), sel=l>>4
// B (32x16 f16) from [N][K]-major LDS: lane l -> row n=l&15; halves [k0+sel*8, +8) and [k0+16+sel*8, +8)
// C (16x16 f32): VGPR r -> (m = r + (l>>4)*8, n = l&15)
// ---------------------------------------------------------------------------
__device__ __forceinline__ v16h load_frag_a(const _Float16* s, int ld, int r0, int k0) {
  int l = threadIdx.x & 31;
  const _Float16* p = s + (size_t)(r0 + (l & 15)) * ld + k0 + (l >> 4) * 8;
  union { v16h v; v8h q[2]; } u;
  u.q[0] = *(const v8h*)p;
  u.q[1] = *(const v8h*)(p + 16);
  return u.v;
}

__device__ __forceinline__ v16h load_frag_b(const _Float16* s, int ld, int n0, int k0) {
  int l = threadIdx.x & 31;
  const _Float16* p = s + (size_t)(n0 + (l & 15)) * ld + k0 + (l >> 4) * 8;
  union { v16h v; v8h q[2]; } u;
  u.q[0] = *(const v8h*)p;
  u.q[1] = *(const v8h*)(p + 16);
  return u.v;
}

__device__ __forceinline__ v8f wmma_f16(v16h a, v16h b, v8f c) {
  return __builtin_amdgcn_wmma_f32_16x16x32_f16(false, a, false, b, (short)0, c, false, false);
}

// ---------------------------------------------------------------------------
// Tensor Data Mover: 2D f16 tile (tile_w x tile_h elements) from global to LDS.
// D# per CDNA5 ISA:
//   group0: count=1 | lds_addr | global_addr[56:0] | type=2
//   group1: data_size=1(2B), pad_enable=1, pad_interval=3 (16 DW = 64B rows),
//           pad_amount=3 (4 DW = 16B) -> reproduces the 40-half LDS row stride.
//           tensor_dim0 [79:48], tensor_dim1 [111:80], tile_dim0 [127:112],
//           tile_dim1 [143:128], tensor_dim0_stride [207:160].
// ---------------------------------------------------------------------------
__device__ __forceinline__ void tdm_load_2d_f16(uint32_t lds_off, const void* gptr,
                                                uint32_t tensor_w, uint32_t tensor_h,
                                                uint32_t stride_elems,
                                                uint32_t tile_w, uint32_t tile_h) {
  uint64_t ga = (uint64_t)(uintptr_t)gptr;
  u32x4 g0;
  g0.x = 1u;                                            // count=1 (user descriptor)
  g0.y = lds_off;                                       // lds_addr (bytes)
  g0.z = (uint32_t)(ga & 0xffffffffu);                  // global_addr[31:0] -> bits 95:64
  g0.w = (uint32_t)((ga >> 32) & 0x01ffffffu) | (2u << 30); // global_addr[56:32], type=2
  i32x8 g1;
  g1[0] = (int)0x06D10000u;                             // mask=0, data_size=1(2B), pad cfg
  g1[1] = (int)((tensor_w & 0xffffu) << 16);            // tensor_dim0 lo -> [63:48]
  g1[2] = (int)(((tensor_w >> 16) & 0xffffu) | ((tensor_h & 0xffffu) << 16));
  g1[3] = (int)(((tensor_h >> 16) & 0xffffu) | ((tile_w & 0xffffu) << 16)); // tile_dim0
  g1[4] = (int)(tile_h & 0xffffu);                      // tile_dim1 (tile_dim2 = 0)
  g1[5] = (int)stride_elems;                            // tensor_dim0_stride[31:0]
  g1[6] = 0;
  g1[7] = 0;
  i32x4 z4 = {0, 0, 0, 0};
#if defined(__clang_major__) && (__clang_major__ >= 23)
  i32x8 z8 = {0, 0, 0, 0, 0, 0, 0, 0};
  __builtin_amdgcn_tensor_load_to_lds(g0, g1, z4, z4, z8, 0);
#else
  __builtin_amdgcn_tensor_load_to_lds(g0, g1, z4, z4, 0);
#endif
}

// ---------------------------------------------------------------------------
// Kernel 1a: tok_e = mean_T(word_emb[tokens]) @ W_tok  -> emb row 0 (f16)
// ---------------------------------------------------------------------------
__global__ __launch_bounds__(128) void tok_embed(const int* __restrict__ tokens,
                                                 const float* __restrict__ wemb,
                                                 const float* __restrict__ Wtok,
                                                 _Float16* __restrict__ emb_h) {
  __shared__ float meanv[304];
  const int b = blockIdx.x, tid = threadIdx.x;
  for (int d = tid; d < 300; d += 128) {
    float acc = 0.f;
    for (int t = 0; t < 20; ++t)
      acc += wemb[(size_t)tokens[b * 20 + t] * 300 + d];
    meanv[d] = acc * (1.0f / 20.0f);
  }
  __syncthreads();
  float acc = 0.f;
  for (int d = 0; d < 300; ++d)
    acc += meanv[d] * Wtok[d * 128 + tid];
  emb_h[(size_t)b * 4096 + tid] = (_Float16)acc;
}

// ---------------------------------------------------------------------------
// Kernel 1b: field embedding gather -> emb rows 1..31 (f16)
// ---------------------------------------------------------------------------
__global__ __launch_bounds__(128) void field_embed(const int* __restrict__ fids,
                                                   const float* __restrict__ ftab,
                                                   _Float16* __restrict__ emb_h) {
  const int b = blockIdx.x, f = blockIdx.y, e = threadIdx.x;
  const int id = fids[b * 31 + f];
  emb_h[(size_t)b * 4096 + (f + 1) * 128 + e] =
      (_Float16)ftab[((size_t)f * 1000 + id) * 128 + e];
}

// ---------------------------------------------------------------------------
// Kernel 2: one full fused self-attention layer. Block = 2 batch rows (M=64).
// ---------------------------------------------------------------------------
__global__ __launch_bounds__(256) void attn_layer_kernel(
    const _Float16* __restrict__ xin, _Float16* __restrict__ xout,
    const float* __restrict__ Wq, const float* __restrict__ Wk,
    const float* __restrict__ Wv, const float* __restrict__ Wr,
    const float* __restrict__ lng, const float* __restrict__ lnb) {
  extern __shared__ char smem[];
  _Float16* Qs  = (_Float16*)smem;            // 64 x LDW
  _Float16* Ks  = Qs + 64 * LDW;              // 64 x LDW
  _Float16* Rs  = Ks + 64 * LDW;              // 64 x LDW
  _Float16* Vts = Rs + 64 * LDW;              // 2 x 128 x VLD (V transposed [d][key])
  char* ovl = smem + 72704;                   // overlay region
  _Float16* Xs = (_Float16*)ovl;              // 64 x LDW
  _Float16* Ws = Xs + 64 * LDW;               // 128 x LDW (weights, [N][K])
  float*    Ss = (float*)ovl;                 // 4 x 32 x 36 (scores)
  _Float16* Ps = (_Float16*)(ovl + 4 * 32 * 36 * 4);              // 4 x 32 x VLD
  float*    Os = (float*)(ovl + 4 * 32 * 36 * 4 + 4 * 32 * VLD * 2); // 64 x 132

  const int tid = threadIdx.x;
  const int lane = tid & 31;
  const int wave = tid >> 5;
  const int hs = lane >> 4, nl = lane & 15;
  const size_t xbase = (size_t)blockIdx.x * 64 * 128;

  // stage X tile (64 rows x 128 halves)
  for (int i = tid; i < 64 * 16; i += 256) {
    int r = i >> 4, c = (i & 15) * 8;
    *(v8h*)&Xs[r * LDW + c] = *(const v8h*)&xin[xbase + r * 128 + c];
  }
  __syncthreads();

  // ---- projections Q,K,V,Res : [64x128] = X[64x128] @ W[128x128]
  for (int w = 0; w < 4; ++w) {
    const float* Wg = (w == 0) ? Wq : (w == 1) ? Wk : (w == 2) ? Wv : Wr;
    for (int i = tid; i < 128 * 128; i += 256) {       // stage W^T f32->f16
      int k = i >> 7, n = i & 127;
      Ws[n * LDW + k] = (_Float16)Wg[i];
    }
    __syncthreads();
    for (int t = 0; t < 4; ++t) {
      int tt = wave * 4 + t;
      int tm = tt >> 3, tn = tt & 7;
      v8f c = {};
#pragma unroll
      for (int kk = 0; kk < 4; ++kk) {
        v16h a = load_frag_a(Xs, LDW, tm * 16, kk * 32);
        v16h b = load_frag_b(Ws, LDW, tn * 16, kk * 32);
        c = wmma_f16(a, b, c);
      }
      if (w == 2) {                                    // V stored transposed [d][key]
#pragma unroll
        for (int r = 0; r < 8; ++r) {
          int m = tm * 16 + r + hs * 8;
          int n = tn * 16 + nl;
          Vts[(m >> 5) * 128 * VLD + n * VLD + (m & 31)] = (_Float16)c[r];
        }
      } else {
        _Float16* dst = (w == 0) ? Qs : (w == 1) ? Ks : Rs;
#pragma unroll
        for (int r = 0; r < 8; ++r)
          dst[(tm * 16 + r + hs * 8) * LDW + tn * 16 + nl] = (_Float16)c[r];
      }
    }
    __syncthreads();
  }

  // ---- S = Q K^T / sqrt(D) : per (batch,head) 32x32, K=64
  for (int t = 0; t < 2; ++t) {
    int tt = wave * 2 + t;
    int hp = tt >> 2, loc = tt & 3, tm = loc >> 1, tn = loc & 1;
    int bi = hp >> 1, h = hp & 1;
    v8f c = {};
#pragma unroll
    for (int kk = 0; kk < 2; ++kk) {
      v16h a = load_frag_a(Qs, LDW, bi * 32 + tm * 16, h * 64 + kk * 32);
      v16h b = load_frag_b(Ks, LDW, bi * 32 + tn * 16, h * 64 + kk * 32);
      c = wmma_f16(a, b, c);
    }
#pragma unroll
    for (int r = 0; r < 8; ++r)
      Ss[hp * 32 * 36 + (tm * 16 + r + hs * 8) * 36 + tn * 16 + nl] = c[r] * 0.125f;
  }
  __syncthreads();

  // ---- row softmax, write P as f16
  if (tid < 128) {
    int hp = tid >> 5, q = tid & 31;
    float* row = &Ss[hp * 32 * 36 + q * 36];
    float mx = -1e30f;
#pragma unroll
    for (int k = 0; k < 32; ++k) mx = fmaxf(mx, row[k]);
    float e[32], s = 0.f;
#pragma unroll
    for (int k = 0; k < 32; ++k) { e[k] = expf(row[k] - mx); s += e[k]; }
    float inv = 1.f / s;
    _Float16* pr = &Ps[hp * 32 * VLD + q * VLD];
#pragma unroll
    for (int k = 0; k < 32; ++k) pr[k] = (_Float16)(e[k] * inv);
  }
  __syncthreads();

  // ---- O = P @ V : per head M=32, N=64, K=32 (single WMMA K-step)
  for (int t = 0; t < 4; ++t) {
    int tt = wave * 4 + t;
    int hp = tt >> 3, loc = tt & 7, tm = loc >> 2, tn = loc & 3;
    int bi = hp >> 1, h = hp & 1;
    v16h a = load_frag_a(Ps + hp * 32 * VLD, VLD, tm * 16, 0);
    v16h b = load_frag_b(Vts + bi * 128 * VLD, VLD, h * 64 + tn * 16, 0);
    v8f c = {};
    c = wmma_f16(a, b, c);
#pragma unroll
    for (int r = 0; r < 8; ++r)
      Os[(bi * 32 + tm * 16 + r + hs * 8) * 132 + h * 64 + tn * 16 + nl] = c[r];
  }
  __syncthreads();

  // ---- residual + ReLU + LayerNorm -> xout (f16)
  if (tid < 64) {
    const int row = tid;
    float s = 0.f, sq = 0.f;
    for (int e0 = 0; e0 < 128; ++e0) {
      float v = Os[row * 132 + e0] + (float)Rs[row * LDW + e0];
      v = fmaxf(v, 0.f);
      s += v; sq += v * v;
    }
    float mu = s * (1.f / 128.f);
    float var = sq * (1.f / 128.f) - mu * mu;
    float rstd = rsqrtf(var + 1e-5f);
    for (int e0 = 0; e0 < 128; ++e0) {
      float v = fmaxf(Os[row * 132 + e0] + (float)Rs[row * LDW + e0], 0.f);
      xout[xbase + row * 128 + e0] = (_Float16)((v - mu) * rstd * lng[e0] + lnb[e0]);
    }
  }
}

// ---------------------------------------------------------------------------
// Kernel 3: tiled WMMA GEMM  C[M,N] = A_f16[M,K] @ W_f32[K,N]  + fused epilogue
// A tile staged by the Tensor Data Mover (async, TENSORcnt), W tile transposed
// f32->f16 by the VALU while the TDM runs.
// BNRELU=true : relu((c - m) * rsqrt(v+eps) * g + b) -> f16
// BNRELU=false: c + bias -> f32
// ---------------------------------------------------------------------------
template <bool BNRELU>
__global__ __launch_bounds__(256) void gemm_kernel(
    const _Float16* __restrict__ A, const float* __restrict__ W,
    _Float16* __restrict__ outh, float* __restrict__ outf,
    const float* __restrict__ g, const float* __restrict__ bt,
    const float* __restrict__ mn, const float* __restrict__ vr,
    const float* __restrict__ bias, int M, int N, int K) {
  __shared__ _Float16 As[128 * VLD];
  __shared__ _Float16 Bs[128 * VLD];
  const int tid = threadIdx.x, lane = tid & 31, wave = tid >> 5;
  const int hs = lane >> 4, nl = lane & 15;
  const int bm = blockIdx.x * 128, bn = blockIdx.y * 128;
  const int wm = (wave & 3) * 32, wn = (wave >> 2) * 64;
  const uint32_t as_off = (uint32_t)(uintptr_t)As;   // LDS byte offset of As
  v8f c[2][4] = {};

  for (int k0 = 0; k0 < K; k0 += 32) {
    // async TDM: A tile (32 x 128 rows, f16) -> As with 16B/row LDS padding
    if (wave == 0)
      tdm_load_2d_f16(as_off, &A[(size_t)bm * K + k0],
                      (uint32_t)(K - k0), 128u, (uint32_t)K, 32u, 128u);
    // stage W^T tile f32->f16 (VALU work overlaps the TDM)
    for (int i = tid; i < 4096; i += 256) {
      int k = i >> 7, n = i & 127;
      Bs[n * VLD + k] = (_Float16)W[(size_t)(k0 + k) * N + bn + n];
    }
    if (k0 + 32 < K)                                  // prefetch next W tile
      __builtin_prefetch(&W[(size_t)(k0 + 32) * N + bn + tid], 0, 0);
    __builtin_amdgcn_s_wait_tensorcnt(0);
    __syncthreads();

    v16h a0 = load_frag_a(As, VLD, wm, 0);
    v16h a1 = load_frag_a(As, VLD, wm + 16, 0);
#pragma unroll
    for (int j = 0; j < 4; ++j) {
      v16h b = load_frag_b(Bs, VLD, wn + j * 16, 0);
      c[0][j] = wmma_f16(a0, b, c[0][j]);
      c[1][j] = wmma_f16(a1, b, c[1][j]);
    }
    __syncthreads();
  }

#pragma unroll
  for (int i = 0; i < 2; ++i)
#pragma unroll
    for (int j = 0; j < 4; ++j)
#pragma unroll
      for (int r = 0; r < 8; ++r) {
        int m = bm + wm + i * 16 + r + hs * 8;
        int n = bn + wn + j * 16 + nl;
        float v = c[i][j][r];
        if constexpr (BNRELU) {
          v = (v - mn[n]) * rsqrtf(vr[n] + 1e-5f) * g[n] + bt[n];
          outh[(size_t)m * N + n] = (_Float16)fmaxf(v, 0.f);
        } else {
          outf[(size_t)m * N + n] = v + bias[n];
        }
      }
}

// ---------------------------------------------------------------------------
// Kernel 4: out[b] = sigmoid( [att | h3] . W_last + b_last )
// ---------------------------------------------------------------------------
__global__ __launch_bounds__(256) void final_out(const _Float16* __restrict__ att,
                                                 const float* __restrict__ h3,
                                                 const float* __restrict__ Wl,
                                                 const float* __restrict__ bl,
                                                 float* __restrict__ out) {
  __shared__ float red[256];
  const int b = blockIdx.x, tid = threadIdx.x;
  float acc = 0.f;
  for (int i = tid; i < 4096; i += 256)
    acc += (float)att[(size_t)b * 4096 + i] * Wl[i];
  if (tid < 128) acc += h3[(size_t)b * 128 + tid] * Wl[4096 + tid];
  red[tid] = acc;
  __syncthreads();
  for (int s = 128; s > 0; s >>= 1) {
    if (tid < s) red[tid] += red[tid + s];
    __syncthreads();
  }
  if (tid == 0) {
    float z = red[0] + bl[0];
    out[b] = 1.0f / (1.0f + expf(-z));
  }
}

// ---------------------------------------------------------------------------
extern "C" void kernel_launch(void* const* d_in, const int* in_sizes, int n_in,
                              void* d_out, int out_size, void* d_ws, size_t ws_size,
                              hipStream_t stream) {
  (void)in_sizes; (void)n_in; (void)out_size; (void)ws_size;
  const int*   tokens = (const int*)  d_in[0];
  const int*   fids   = (const int*)  d_in[1];
  const float* wemb   = (const float*)d_in[2];
  const float* Wtok   = (const float*)d_in[3];
  const float* ftab   = (const float*)d_in[4];
  const float* Wq     = (const float*)d_in[5];
  const float* Wk     = (const float*)d_in[6];
  const float* Wv     = (const float*)d_in[7];
  const float* Wr     = (const float*)d_in[8];
  const float* lng    = (const float*)d_in[9];
  const float* lnb    = (const float*)d_in[10];
  const float* W1     = (const float*)d_in[11];
  const float* bn1g   = (const float*)d_in[12];
  const float* bn1b   = (const float*)d_in[13];
  const float* bn1m   = (const float*)d_in[14];
  const float* bn1v   = (const float*)d_in[15];
  const float* W2     = (const float*)d_in[16];
  const float* bn2g   = (const float*)d_in[17];
  const float* bn2b   = (const float*)d_in[18];
  const float* bn2m   = (const float*)d_in[19];
  const float* bn2v   = (const float*)d_in[20];
  const float* W3     = (const float*)d_in[21];
  const float* b3     = (const float*)d_in[22];
  const float* Wlast  = (const float*)d_in[23];
  const float* blast  = (const float*)d_in[24];
  float* out = (float*)d_out;

  char* ws = (char*)d_ws;
  _Float16* emb_h = (_Float16*)(ws);               // 4096*4096 f16 = 32 MiB
  _Float16* x_h   = (_Float16*)(ws + 33554432);    // 32 MiB
  _Float16* h1    = (_Float16*)(ws + 67108864);    // 4096*1024 f16 = 8 MiB
  _Float16* h2    = (_Float16*)(ws + 75497472);    // 4096*512  f16 = 4 MiB
  float*    h3    = (float*)   (ws + 79691776);    // 4096*128  f32 = 2 MiB

  // 1) embeddings
  tok_embed<<<4096, 128, 0, stream>>>(tokens, wemb, Wtok, emb_h);
  field_embed<<<dim3(4096, 31), 128, 0, stream>>>(fids, ftab, emb_h);

  // 2) attention stack (dynamic LDS: 72704 persistent + 62464 overlay)
  constexpr size_t SM = 72704 + 62464;
  for (int l = 0; l < 3; ++l) {
    const _Float16* xi = (l == 0) ? emb_h : x_h;
    attn_layer_kernel<<<2048, 256, SM, stream>>>(
        xi, x_h, Wq + l * 16384, Wk + l * 16384, Wv + l * 16384, Wr + l * 16384,
        lng + l * 128, lnb + l * 128);
  }

  // 3) DNN tower (TDM-staged A tiles)
  gemm_kernel<true><<<dim3(32, 8), 256, 0, stream>>>(
      emb_h, W1, h1, nullptr, bn1g, bn1b, bn1m, bn1v, nullptr, 4096, 1024, 4096);
  gemm_kernel<true><<<dim3(32, 4), 256, 0, stream>>>(
      h1, W2, h2, nullptr, bn2g, bn2b, bn2m, bn2v, nullptr, 4096, 512, 1024);
  gemm_kernel<false><<<dim3(32, 1), 256, 0, stream>>>(
      h2, W3, nullptr, h3, nullptr, nullptr, nullptr, nullptr, b3, 4096, 128, 512);

  // 4) head
  final_out<<<4096, 256, 0, stream>>>(x_h, h3, Wlast, blast, out);
}
